// MambaBlock_1846835938126
// MI455X (gfx1250) — compile-verified, matrix operand order checked by reading, not run on hardware
//
#include <hip/hip_runtime.h>
#include <hip/hip_bf16.h>

// ---------------------------------------------------------------------------
// Types for CDNA5 WMMA (gfx1250, wave32)
// ---------------------------------------------------------------------------
typedef __bf16 bf16_t;
typedef float  v8f  __attribute__((ext_vector_type(8)));
typedef bf16_t v16bf __attribute__((ext_vector_type(16)));
typedef bf16_t v8bf  __attribute__((ext_vector_type(8)));
typedef int    v4i_t __attribute__((ext_vector_type(4)));

// Model dims
#define Bdim 4
#define Tdim 2048
#define Ddim 512
#define Hdim 1024          // d_inner
#define H2   2048          // 2H
#define H3   3072          // 3H
#define MROWS 8192         // B*T
#define NWG_GRU 64         // persistent workgroups for the recurrence

// ---------------------------------------------------------------------------
// gfx1250 async Global->LDS copy (ASYNCcnt-tracked); vector-copy fallback.
// Builtin signature (from hipcc diagnostic): (v4i AS1* src, v4i AS3* dst,
// imm offset, imm cpol).
// ---------------------------------------------------------------------------
#if __has_builtin(__builtin_amdgcn_global_load_async_to_lds_b128)
#define HAVE_ASYNC_LDS 1
#else
#define HAVE_ASYNC_LDS 0
#endif

__device__ __forceinline__ void async_copy16(const bf16_t* gsrc, bf16_t* ldst) {
#if HAVE_ASYNC_LDS
    __builtin_amdgcn_global_load_async_to_lds_b128(
        (__attribute__((address_space(1))) v4i_t*)gsrc,
        (__attribute__((address_space(3))) v4i_t*)ldst, 0, 0);
#else
    *reinterpret_cast<v8bf*>(ldst) = *reinterpret_cast<const v8bf*>(gsrc);
#endif
}

__device__ __forceinline__ void wait_async0() {
#if HAVE_ASYNC_LDS
#if __has_builtin(__builtin_amdgcn_s_wait_asynccnt)
    __builtin_amdgcn_s_wait_asynccnt(0);
#else
    asm volatile("s_wait_asynccnt 0" ::: "memory");
#endif
#endif
}

// ---------------------------------------------------------------------------
// Fragment loader: per ISA 7.12.2, a 16-bit A/B fragment for 16x16x32 is,
// for lane L (m/n = L%16, h = L/16), two contiguous 8-element (16B) chunks:
//   vector elems [0..7]  = row[kb + h*8 + 0..7]
//   vector elems [8..15] = row[kb + 16 + h*8 + 0..7]
// Works for both global and LDS (generic pointer -> ds_load_b128).
// ---------------------------------------------------------------------------
__device__ __forceinline__ v16bf load_frag(const bf16_t* __restrict__ row, int kb, int h) {
    v8bf lo = *reinterpret_cast<const v8bf*>(row + kb + h * 8);
    v8bf hi = *reinterpret_cast<const v8bf*>(row + kb + 16 + h * 8);
    return __builtin_shufflevector(lo, hi, 0,1,2,3,4,5,6,7,8,9,10,11,12,13,14,15);
}

__device__ __forceinline__ float fsigmoid(float x) { return 1.0f / (1.0f + __expf(-x)); }

// ---------------------------------------------------------------------------
// fp32 -> bf16 elementwise convert (for weights)
// ---------------------------------------------------------------------------
__global__ void k_f2bf(const float* __restrict__ in, bf16_t* __restrict__ out, int n) {
    int i = blockIdx.x * 256 + threadIdx.x;
    if (i < n) out[i] = (bf16_t)in[i];
}

// ---------------------------------------------------------------------------
// LayerNorm over D=512, output bf16. One block (256 threads) per row.
// ---------------------------------------------------------------------------
__global__ void k_layernorm(const float* __restrict__ x, const float* __restrict__ g,
                            const float* __restrict__ b, bf16_t* __restrict__ out) {
    __shared__ float s1[256];
    __shared__ float s2[256];
    const int row = blockIdx.x;
    const int tid = threadIdx.x;
    const float* xr = x + (size_t)row * Ddim;
    float a0 = xr[tid], a1 = xr[tid + 256];
    s1[tid] = a0 + a1;
    s2[tid] = a0 * a0 + a1 * a1;
    __syncthreads();
    for (int s = 128; s > 0; s >>= 1) {
        if (tid < s) { s1[tid] += s1[tid + s]; s2[tid] += s2[tid + s]; }
        __syncthreads();
    }
    const float mean = s1[0] * (1.0f / Ddim);
    const float var  = s2[0] * (1.0f / Ddim) - mean * mean;
    const float rstd = rsqrtf(var + 1e-5f);
    out[(size_t)row * Ddim + tid]       = (bf16_t)((a0 - mean) * rstd * g[tid] + b[tid]);
    out[(size_t)row * Ddim + tid + 256] = (bf16_t)((a1 - mean) * rstd * g[tid + 256] + b[tid + 256]);
}

// ---------------------------------------------------------------------------
// bf16 WMMA GEMM: C(MxN,f32) = A(MxK,bf16) * W(NxK,bf16)^T + bias [+ resid]
// Block = 256 threads = 8 waves. The block's 16xK A strip (contiguous,
// row-major) is async-staged into LDS once; each wave then computes FOUR
// 16x16 output tiles (independent WMMA chains) reusing the A fragment.
// grid.x = N/512, grid.y = M/16.  Requires K <= 1024, K % 64 == 0.
// ---------------------------------------------------------------------------
__global__ void k_gemm_bf16(const bf16_t* __restrict__ A, const bf16_t* __restrict__ W,
                            const float* __restrict__ bias, const float* __restrict__ resid,
                            float* __restrict__ C, int M, int N, int K) {
    __shared__ bf16_t aTile[16 * 1024];       // up to 32KB of the 320KB WGP LDS
    const int tid  = threadIdx.x;
    const int lane = tid & 31;
    const int wave = tid >> 5;
    const int l16  = lane & 15;
    const int hh   = lane >> 4;
    const int m0    = blockIdx.y * 16;
    const int nbase = blockIdx.x * 512 + wave * 64;

    // stage the contiguous 16xK A strip into LDS (ASYNCcnt path on gfx1250)
    const bf16_t* astrip = A + (size_t)m0 * K;
    const int chunks = (16 * K) >> 3;         // 8 bf16 = 16B per chunk
    for (int c = tid; c < chunks; c += 256)
        async_copy16(astrip + (size_t)c * 8, aTile + c * 8);
    wait_async0();
    __syncthreads();

    const bf16_t* Arow = aTile + l16 * K;
    const bf16_t* W0 = W + (size_t)(nbase +  0 + l16) * K;
    const bf16_t* W1 = W + (size_t)(nbase + 16 + l16) * K;
    const bf16_t* W2 = W + (size_t)(nbase + 32 + l16) * K;
    const bf16_t* W3 = W + (size_t)(nbase + 48 + l16) * K;

    v8f acc0 = {}, acc1 = {}, acc2 = {}, acc3 = {};
#pragma unroll 2
    for (int kb = 0; kb < K; kb += 32) {
        v16bf av = load_frag(Arow, kb, hh);
        v16bf b0 = load_frag(W0, kb, hh);
        v16bf b1 = load_frag(W1, kb, hh);
        v16bf b2 = load_frag(W2, kb, hh);
        v16bf b3 = load_frag(W3, kb, hh);
        acc0 = __builtin_amdgcn_wmma_f32_16x16x32_bf16(false, av, false, b0, (short)0, acc0, false, false);
        acc1 = __builtin_amdgcn_wmma_f32_16x16x32_bf16(false, av, false, b1, (short)0, acc1, false, false);
        acc2 = __builtin_amdgcn_wmma_f32_16x16x32_bf16(false, av, false, b2, (short)0, acc2, false, false);
        acc3 = __builtin_amdgcn_wmma_f32_16x16x32_bf16(false, av, false, b3, (short)0, acc3, false, false);
    }

#pragma unroll
    for (int j = 0; j < 4; ++j) {
        const v8f acc = (j == 0) ? acc0 : (j == 1) ? acc1 : (j == 2) ? acc2 : acc3;
        const int n = nbase + j * 16 + l16;
        const float bn = bias ? bias[n] : 0.0f;
#pragma unroll
        for (int r = 0; r < 8; ++r) {
            const int m = m0 + r + 8 * hh;   // C/D layout: VGPR r -> row r + 8*(lane/16)
            float v = acc[r] + bn;
            if (resid) v += resid[(size_t)m * N + n];
            C[(size_t)m * N + n] = v;
        }
    }
}

// ---------------------------------------------------------------------------
// Causal depthwise conv1d (k=4) over x_proj = xz[:, :H], + bias, SiLU -> bf16
// ---------------------------------------------------------------------------
__global__ void k_conv_silu(const float* __restrict__ xz, const float* __restrict__ cw,
                            const float* __restrict__ cb, bf16_t* __restrict__ xconv) {
    const size_t idx = (size_t)blockIdx.x * 256 + threadIdx.x;   // B*T*H
    const int    c   = (int)(idx & (Hdim - 1));
    const size_t m   = idx >> 10;                                 // b*T + t
    const int    t   = (int)(m & (Tdim - 1));
    const size_t bb  = m >> 11;
    float acc = cb[c];
#pragma unroll
    for (int k = 0; k < 4; ++k) {
        const int tt = t + k - 3;
        if (tt >= 0) acc += xz[((bb * Tdim + tt) * (size_t)H2) + c] * cw[c * 4 + k];
    }
    xconv[idx] = (bf16_t)(acc * fsigmoid(acc));
}

// ---------------------------------------------------------------------------
// Persistent GRU recurrence. 64 WGs x 128 threads. WG wg owns h cols
// [wg*16, wg*16+16). Each step: async-stage the (zero-padded) 16x1024 bf16
// h matrix into LDS, waves 0..2 compute gate r/z/n slices of
// hh = h @ w_hh^T + b_hh with two independent WMMA chains over K, then 64
// threads do the elementwise update plus the fused output y = h_new*silu(z).
// Grid-wide step sync: monotonic atomic counter in d_ws.
// ---------------------------------------------------------------------------
__global__ void k_gru(const float* __restrict__ pre, const float* __restrict__ xz,
                      const float* __restrict__ b_hh, const bf16_t* __restrict__ w_hh,
                      float* __restrict__ h_f32, bf16_t* __restrict__ h_bf,
                      bf16_t* __restrict__ y, unsigned int* __restrict__ bar) {
    __shared__ bf16_t hTile[16 * Hdim];      // 32KB staged h (rows 4..15 zero)
    __shared__ float  lds_hh[3 * 64];        // [gate][batch][col]
    const int tid  = threadIdx.x;
    const int lane = tid & 31;
    const int wave = tid >> 5;
    const int l16  = lane & 15;
    const int hh   = lane >> 4;
    const int colb = blockIdx.x * 16;        // this WG's h-column base

    for (int t = 0; t < Tdim; ++t) {
        // stage h (16x1024 bf16, contiguous) into LDS
        for (int c = tid; c < (16 * Hdim) >> 3; c += 128)
            async_copy16(h_bf + (size_t)c * 8, hTile + c * 8);
        wait_async0();
        __syncthreads();

        if (wave < 3) {                       // wave-uniform: EXEC all-ones for WMMA
            const int g = wave;
            const bf16_t* Wrow = w_hh + (size_t)(g * Hdim + colb + l16) * Hdim;
            const bf16_t* Arow = hTile + l16 * Hdim;
            v8f accA = {}, accB = {};
#pragma unroll 2
            for (int kb = 0; kb < Hdim / 2; kb += 32) {   // two independent chains
                v16bf a0 = load_frag(Arow, kb, hh);
                v16bf b0 = load_frag(Wrow, kb, hh);
                v16bf a1 = load_frag(Arow, kb + Hdim / 2, hh);
                v16bf b1 = load_frag(Wrow, kb + Hdim / 2, hh);
                accA = __builtin_amdgcn_wmma_f32_16x16x32_bf16(false, a0, false, b0, (short)0, accA, false, false);
                accB = __builtin_amdgcn_wmma_f32_16x16x32_bf16(false, a1, false, b1, (short)0, accB, false, false);
            }
            const v8f acc = accA + accB;
            const float bb = b_hh[g * Hdim + colb + l16];
#pragma unroll
            for (int r = 0; r < 8; ++r) {
                const int m = r + 8 * hh;
                if (m < Bdim) lds_hh[g * 64 + m * 16 + l16] = acc[r] + bb;
            }
        }
        __syncthreads();
        if (tid < 64) {
            const int b = tid >> 4, j = tid & 15;
            const int col = colb + j;
            const size_t rowp = ((size_t)b * Tdim + t) * H3;
            const float pr = pre[rowp + col];
            const float pz = pre[rowp + Hdim + col];
            const float pn = pre[rowp + 2 * Hdim + col];
            const float hr = lds_hh[b * 16 + j];
            const float hz = lds_hh[64 + b * 16 + j];
            const float hn = lds_hh[128 + b * 16 + j];
            const float hold = h_f32[b * Hdim + col];
            const float rg = fsigmoid(pr + hr);
            const float zg = fsigmoid(pz + hz);
            const float ng = tanhf(pn + rg * hn);
            const float hnew = (1.0f - zg) * ng + zg * hold;
            h_f32[b * Hdim + col] = hnew;
            h_bf[b * Hdim + col]  = (bf16_t)hnew;
            const float zv = xz[((size_t)b * Tdim + t) * H2 + Hdim + col];
            y[((size_t)b * Tdim + t) * Hdim + col] = (bf16_t)(hnew * (zv * fsigmoid(zv)));
        }
        __syncthreads();
        if (tid == 0) {
            __hip_atomic_fetch_add(bar, 1u, __ATOMIC_RELEASE, __HIP_MEMORY_SCOPE_AGENT);
            const unsigned target = (unsigned)NWG_GRU * (unsigned)(t + 1);
            while (__hip_atomic_load(bar, __ATOMIC_ACQUIRE, __HIP_MEMORY_SCOPE_AGENT) < target)
                __builtin_amdgcn_s_sleep(1);
        }
        __syncthreads();
    }
}

// ---------------------------------------------------------------------------
// Host-side launcher
// ---------------------------------------------------------------------------
extern "C" void kernel_launch(void* const* d_in, const int* in_sizes, int n_in,
                              void* d_out, int out_size, void* d_ws, size_t ws_size,
                              hipStream_t stream) {
    (void)in_sizes; (void)n_in; (void)out_size; (void)ws_size;
    const float* x      = (const float*)d_in[0];
    const float* ln_g   = (const float*)d_in[1];
    const float* ln_b   = (const float*)d_in[2];
    const float* in_w   = (const float*)d_in[3];
    const float* in_b   = (const float*)d_in[4];
    const float* conv_w = (const float*)d_in[5];
    const float* conv_b = (const float*)d_in[6];
    const float* w_ih   = (const float*)d_in[7];
    const float* w_hh   = (const float*)d_in[8];
    const float* b_ih   = (const float*)d_in[9];
    const float* b_hh   = (const float*)d_in[10];
    const float* out_w  = (const float*)d_in[11];
    const float* out_b  = (const float*)d_in[12];
    float* out = (float*)d_out;

    char* ws = (char*)d_ws;
    size_t off = 0;
    auto alloc = [&](size_t bytes) -> void* {
        void* p = ws + off;
        off = (off + bytes + 255) & ~(size_t)255;
        return p;
    };
    // ---- zeroed head: barrier counter + GRU state (h fp32 4x1024, h bf16 16x1024 padded)
    unsigned* bar   = (unsigned*)alloc(sizeof(unsigned));
    float*    h_f32 = (float*)   alloc((size_t)Bdim * Hdim * sizeof(float));
    bf16_t*   h_bf  = (bf16_t*)  alloc((size_t)16 * Hdim * sizeof(bf16_t));
    const size_t zero_bytes = off;
    // ---- the rest of the workspace
    bf16_t* xn_bf    = (bf16_t*)alloc((size_t)MROWS * Ddim * sizeof(bf16_t));
    bf16_t* in_w_bf  = (bf16_t*)alloc((size_t)H2 * Ddim * sizeof(bf16_t));
    bf16_t* w_ih_bf  = (bf16_t*)alloc((size_t)H3 * Hdim * sizeof(bf16_t));
    bf16_t* w_hh_bf  = (bf16_t*)alloc((size_t)H3 * Hdim * sizeof(bf16_t));
    bf16_t* out_w_bf = (bf16_t*)alloc((size_t)Ddim * Hdim * sizeof(bf16_t));
    float*  xz       = (float*) alloc((size_t)MROWS * H2 * sizeof(float));
    bf16_t* xconv    = (bf16_t*)alloc((size_t)MROWS * Hdim * sizeof(bf16_t));
    float*  pre      = (float*) alloc((size_t)MROWS * H3 * sizeof(float));
    bf16_t* y_bf     = (bf16_t*)alloc((size_t)MROWS * Hdim * sizeof(bf16_t));

    (void)hipMemsetAsync(d_ws, 0, zero_bytes, stream);

    // weight conversions fp32 -> bf16
    k_f2bf<<<(H2 * Ddim + 255) / 256, 256, 0, stream>>>(in_w, in_w_bf, H2 * Ddim);
    k_f2bf<<<(H3 * Hdim + 255) / 256, 256, 0, stream>>>(w_ih, w_ih_bf, H3 * Hdim);
    k_f2bf<<<(H3 * Hdim + 255) / 256, 256, 0, stream>>>(w_hh, w_hh_bf, H3 * Hdim);
    k_f2bf<<<(Ddim * Hdim + 255) / 256, 256, 0, stream>>>(out_w, out_w_bf, Ddim * Hdim);

    // 1) LayerNorm -> bf16
    k_layernorm<<<MROWS, 256, 0, stream>>>(x, ln_g, ln_b, xn_bf);

    // 2) in_proj: xz = xn @ in_w^T + in_b   (8192 x 2048, K=512)
    k_gemm_bf16<<<dim3(H2 / 512, MROWS / 16), 256, 0, stream>>>(
        xn_bf, in_w_bf, in_b, nullptr, xz, MROWS, H2, Ddim);

    // 3) causal depthwise conv + SiLU -> bf16
    k_conv_silu<<<(MROWS * Hdim) / 256, 256, 0, stream>>>(xz, conv_w, conv_b, xconv);

    // 4) pre = x_conv @ w_ih^T + b_ih   (8192 x 3072, K=1024)
    k_gemm_bf16<<<dim3(H3 / 512, MROWS / 16), 256, 0, stream>>>(
        xconv, w_ih_bf, b_ih, nullptr, pre, MROWS, H3, Hdim);

    // 5) persistent GRU recurrence + fused y = h * silu(z)
    k_gru<<<NWG_GRU, 128, 0, stream>>>(pre, xz, b_hh, w_hh_bf, h_f32, h_bf, y_bf, bar);

    // 6) out = y @ out_w^T + out_b + x   (8192 x 512, K=1024), residual fused
    k_gemm_bf16<<<dim3(Ddim / 512, MROWS / 16), 256, 0, stream>>>(
        y_bf, out_w_bf, out_b, x, out, MROWS, Ddim, Hdim);
}
